// MaskRCNN_67250597921485
// MI455X (gfx1250) — matrix operand order repaired and checked by reading, not compiled, not run on hardware
//
#include <hip/hip_runtime.h>
#include <hip/hip_bf16.h>
#include <cstddef>
#include <cstdint>

#define USE_ASYNC 1

// ---------------------------------------------------------------------------
// Types / helpers
// ---------------------------------------------------------------------------
typedef __attribute__((ext_vector_type(16))) unsigned short v16u;
typedef __attribute__((ext_vector_type(8)))  float          v8f;
#if defined(__HIP_DEVICE_COMPILE__)
typedef __attribute__((ext_vector_type(16))) __bf16         v16bf;
#endif

__device__ __forceinline__ unsigned short f2bf(float f) {
  unsigned u = __float_as_uint(f);
  unsigned r = (u + 0x7FFFu + ((u >> 16) & 1u)) >> 16;   // RNE
  return (unsigned short)r;
}
__device__ __forceinline__ float bf2f(unsigned short h) {
  return __uint_as_float(((unsigned)h) << 16);
}

__device__ __forceinline__ v8f wmma_bf16_16x16x32(v16u au, v16u bu, v8f c) {
#if defined(__HIP_DEVICE_COMPILE__)
  return __builtin_amdgcn_wmma_f32_16x16x32_bf16(
      false, __builtin_bit_cast(v16bf, au),
      false, __builtin_bit_cast(v16bf, bu),
      (short)0, c, false, false);
#else
  (void)au; (void)bu; return c;
#endif
}

// Async memory->LDS 16B copy (per active lane), tracked by ASYNCcnt.
__device__ __forceinline__ void async_ld_b128(void* lds_ptr, const void* gptr) {
#if defined(__HIP_DEVICE_COMPILE__) && USE_ASYNC
  unsigned lo = (unsigned)(uintptr_t)lds_ptr;   // low 32 bits of generic = LDS offset
  unsigned long long ga = (unsigned long long)(uintptr_t)gptr;
  asm volatile("global_load_async_to_lds_b128 %0, %1, off"
               :: "v"(lo), "v"(ga) : "memory");
#else
  *(uint4*)lds_ptr = *(const uint4*)gptr;
#endif
}
__device__ __forceinline__ void wait_async0() {
#if defined(__HIP_DEVICE_COMPILE__) && USE_ASYNC
  asm volatile("s_wait_asynccnt 0x0" ::: "memory");
#endif
}

// GEMM tiling: block tile 64x64, K-step 32, 128 threads = 4 waves,
// each wave owns a 32x32 quadrant = 2x2 WMMA (16x16) tiles.
#define BMN 64
#define BK  32
#define LDT 40   // padded LDS row stride (halfs); 80B keeps 16B-aligned vectors

// A fragment (16x32 bf16): lanes 0-15 M=lane, K {0..7,16..23}; lanes 16-31 K+8.
__device__ __forceinline__ v16u frag_a(const unsigned short* As, int mr, int half) {
  union { v16u v; uint4 q[2]; } u;
  u.q[0] = *(const uint4*)&As[mr * LDT + 8 * half];
  u.q[1] = *(const uint4*)&As[mr * LDT + 16 + 8 * half];
  return u.v;
}
// B fragment (32x16 bf16) from transposed tile Bt[n][k]:
// lanes 0-15: N=lane, K {0..7,16..23}; lanes 16-31 K+8.
__device__ __forceinline__ v16u frag_b(const unsigned short* Bt, int nc, int half) {
  union { v16u v; uint4 q[2]; } u;
  u.q[0] = *(const uint4*)&Bt[nc * LDT + 8 * half];
  u.q[1] = *(const uint4*)&Bt[nc * LDT + 16 + 8 * half];
  return u.v;
}

__device__ __forceinline__ float iou_f(const float* a, const float* c) {
  float areaA = fmaxf(a[2] - a[0], 0.f) * fmaxf(a[3] - a[1], 0.f);
  float areaB = fmaxf(c[2] - c[0], 0.f) * fmaxf(c[3] - c[1], 0.f);
  float iy1 = fmaxf(a[0], c[0]), ix1 = fmaxf(a[1], c[1]);
  float iy2 = fminf(a[2], c[2]), ix2 = fminf(a[3], c[3]);
  float inter = fmaxf(iy2 - iy1, 0.f) * fmaxf(ix2 - ix1, 0.f);
  return inter / (areaA + areaB - inter + 1e-8f);
}

// ---------------------------------------------------------------------------
// f32 -> bf16 conversion
// ---------------------------------------------------------------------------
__global__ void cvt_f32_bf16_kernel(const float* __restrict__ in,
                                    unsigned short* __restrict__ out, size_t n) {
  size_t i = (size_t)blockIdx.x * blockDim.x + threadIdx.x;
  size_t stride = (size_t)gridDim.x * blockDim.x;
  for (; i < n; i += stride) out[i] = f2bf(in[i]);
}

// ---------------------------------------------------------------------------
// Generic GEMM (bf16 x bf16 -> f32 acc), bias + optional relu, bf16 out.
// M,N,K multiples of 64/64/32. Block = 128 thr, tile 64x64.
// ---------------------------------------------------------------------------
__global__ __launch_bounds__(128)
void gemm_bf16_wmma_kernel(const unsigned short* __restrict__ A,
                           const unsigned short* __restrict__ Bw,
                           const float* __restrict__ bias,
                           unsigned short* __restrict__ Out,
                           int M, int N, int K, int relu) {
  (void)M;
  __shared__ __align__(16) unsigned short As[BMN * LDT];
  __shared__ __align__(16) unsigned short Bt[BMN * LDT];
  const int tid = threadIdx.x;
  const int wave = tid >> 5, lane = tid & 31;
  const int half = lane >> 4;
  const int l15 = lane & 15;
  const int m0 = blockIdx.x * BMN, n0 = blockIdx.y * BMN;
  const int wm0 = (wave >> 1) * 32, wn0 = (wave & 1) * 32;
  // A staging: rows rA0/rA1, 8-half column group
  const int rA0 = tid >> 2, rA1 = 32 + (tid >> 2);
  const int ca8 = (tid & 3) * 8;
  // B staging: k-rows rB0/rB1, 8-half n group (stored transposed)
  const int rB0 = tid >> 3, rB1 = 16 + (tid >> 3);
  const int nb8 = (tid & 7) * 8;
  v8f acc[2][2];
#pragma unroll
  for (int ti = 0; ti < 2; ++ti)
#pragma unroll
    for (int tj = 0; tj < 2; ++tj)
      acc[ti][tj] = (v8f){0.f, 0.f, 0.f, 0.f, 0.f, 0.f, 0.f, 0.f};

  for (int k0 = 0; k0 < K; k0 += BK) {
    // A tile: async memory -> LDS, row-major
    async_ld_b128(&As[rA0 * LDT + ca8], A + (size_t)(m0 + rA0) * K + k0 + ca8);
    async_ld_b128(&As[rA1 * LDT + ca8], A + (size_t)(m0 + rA1) * K + k0 + ca8);
    // B tile: registers -> transposed LDS scatter
    {
      union { uint4 q; unsigned short h[8]; } bu;
      bu.q = *(const uint4*)(Bw + (size_t)(k0 + rB0) * N + n0 + nb8);
#pragma unroll
      for (int e = 0; e < 8; ++e) Bt[(nb8 + e) * LDT + rB0] = bu.h[e];
      bu.q = *(const uint4*)(Bw + (size_t)(k0 + rB1) * N + n0 + nb8);
#pragma unroll
      for (int e = 0; e < 8; ++e) Bt[(nb8 + e) * LDT + rB1] = bu.h[e];
    }
    wait_async0();
    __syncthreads();
    const v16u a0 = frag_a(As, wm0 + l15, half);
    const v16u a1 = frag_a(As, wm0 + 16 + l15, half);
    const v16u b0 = frag_b(Bt, wn0 + l15, half);
    const v16u b1 = frag_b(Bt, wn0 + 16 + l15, half);
    acc[0][0] = wmma_bf16_16x16x32(a0, b0, acc[0][0]);
    acc[0][1] = wmma_bf16_16x16x32(a0, b1, acc[0][1]);
    acc[1][0] = wmma_bf16_16x16x32(a1, b0, acc[1][0]);
    acc[1][1] = wmma_bf16_16x16x32(a1, b1, acc[1][1]);
    __syncthreads();
  }
#pragma unroll
  for (int ti = 0; ti < 2; ++ti)
#pragma unroll
    for (int tj = 0; tj < 2; ++tj) {
      const int n = n0 + wn0 + tj * 16 + l15;
      const float bv = bias[n];
#pragma unroll
      for (int rr = 0; rr < 8; ++rr) {
        const int m = m0 + wm0 + ti * 16 + rr + 8 * half;
        float v = acc[ti][tj][rr] + bv;
        if (relu) v = fmaxf(v, 0.f);
        Out[(size_t)m * N + n] = f2bf(v);
      }
    }
}

// ---------------------------------------------------------------------------
// 3x3 SAME conv as GEMM with on-the-fly im2col (K = 9*256, N = 256).
// feat: bf16 [(b*S+y)*S+x]*256+c ; wconv: bf16 [2304,256]; out hfeat (relu).
// ---------------------------------------------------------------------------
__global__ __launch_bounds__(128)
void conv3x3_wmma_kernel(const unsigned short* __restrict__ feat,
                         const unsigned short* __restrict__ wconv,
                         const float* __restrict__ bias,
                         unsigned short* __restrict__ Out,
                         int S, int Bsz) {
  (void)Bsz;
  __shared__ __align__(16) unsigned short As[BMN * LDT];
  __shared__ __align__(16) unsigned short Bt[BMN * LDT];
  const int tid = threadIdx.x;
  const int wave = tid >> 5, lane = tid & 31;
  const int half = lane >> 4;
  const int l15 = lane & 15;
  const int m0 = blockIdx.x * BMN, n0 = blockIdx.y * BMN;
  const int wm0 = (wave >> 1) * 32, wn0 = (wave & 1) * 32;
  const int rA0 = tid >> 2, rA1 = 32 + (tid >> 2);
  const int ca8 = (tid & 3) * 8;
  const int rB0 = tid >> 3, rB1 = 16 + (tid >> 3);
  const int nb8 = (tid & 7) * 8;
  // decode im2col pixel rows once
  const int mA0 = m0 + rA0, mA1 = m0 + rA1;
  const int bb0 = mA0 / (S * S), rem0 = mA0 % (S * S), yy0 = rem0 / S, xx0 = rem0 % S;
  const int bb1 = mA1 / (S * S), rem1 = mA1 % (S * S), yy1 = rem1 / S, xx1 = rem1 % S;
  v8f acc[2][2];
#pragma unroll
  for (int ti = 0; ti < 2; ++ti)
#pragma unroll
    for (int tj = 0; tj < 2; ++tj)
      acc[ti][tj] = (v8f){0.f, 0.f, 0.f, 0.f, 0.f, 0.f, 0.f, 0.f};

  const int K = 2304;
  const uint4 z4 = {0u, 0u, 0u, 0u};
  for (int k0 = 0; k0 < K; k0 += BK) {
    const int tap = k0 >> 8;            // 0..8; K-tile stays inside one tap
    const int ky = tap / 3 - 1, kx = tap % 3 - 1;
    const int c0 = (k0 & 255) + ca8;
    {
      const int ys = yy0 + ky, xs = xx0 + kx;
      if (ys >= 0 && ys < S && xs >= 0 && xs < S)
        async_ld_b128(&As[rA0 * LDT + ca8],
                      feat + ((size_t)(bb0 * S + ys) * S + xs) * 256 + c0);
      else
        *(uint4*)&As[rA0 * LDT + ca8] = z4;
    }
    {
      const int ys = yy1 + ky, xs = xx1 + kx;
      if (ys >= 0 && ys < S && xs >= 0 && xs < S)
        async_ld_b128(&As[rA1 * LDT + ca8],
                      feat + ((size_t)(bb1 * S + ys) * S + xs) * 256 + c0);
      else
        *(uint4*)&As[rA1 * LDT + ca8] = z4;
    }
    {
      union { uint4 q; unsigned short h[8]; } bu;
      bu.q = *(const uint4*)(wconv + (size_t)(k0 + rB0) * 256 + n0 + nb8);
#pragma unroll
      for (int e = 0; e < 8; ++e) Bt[(nb8 + e) * LDT + rB0] = bu.h[e];
      bu.q = *(const uint4*)(wconv + (size_t)(k0 + rB1) * 256 + n0 + nb8);
#pragma unroll
      for (int e = 0; e < 8; ++e) Bt[(nb8 + e) * LDT + rB1] = bu.h[e];
    }
    wait_async0();
    __syncthreads();
    const v16u a0 = frag_a(As, wm0 + l15, half);
    const v16u a1 = frag_a(As, wm0 + 16 + l15, half);
    const v16u b0 = frag_b(Bt, wn0 + l15, half);
    const v16u b1 = frag_b(Bt, wn0 + 16 + l15, half);
    acc[0][0] = wmma_bf16_16x16x32(a0, b0, acc[0][0]);
    acc[0][1] = wmma_bf16_16x16x32(a0, b1, acc[0][1]);
    acc[1][0] = wmma_bf16_16x16x32(a1, b0, acc[1][0]);
    acc[1][1] = wmma_bf16_16x16x32(a1, b1, acc[1][1]);
    __syncthreads();
  }
#pragma unroll
  for (int ti = 0; ti < 2; ++ti)
#pragma unroll
    for (int tj = 0; tj < 2; ++tj) {
      const int n = n0 + wn0 + tj * 16 + l15;
      const float bv = bias[n];
#pragma unroll
      for (int rr = 0; rr < 8; ++rr) {
        const int mm = m0 + wm0 + ti * 16 + rr + 8 * half;
        float v = fmaxf(acc[ti][tj][rr] + bv, 0.f);
        Out[(size_t)mm * 256 + n] = f2bf(v);
      }
    }
}

// ---------------------------------------------------------------------------
// RPN head: 1x1 cls (A=3) + 1x1 box (12) + sigmoid + anchor decode + clip.
// ---------------------------------------------------------------------------
__global__ void rpn_pred_kernel(const unsigned short* __restrict__ hfeat,
                                const float* __restrict__ wcls, const float* __restrict__ bcls,
                                const float* __restrict__ wbox, const float* __restrict__ bbox,
                                const float* __restrict__ info,
                                float* __restrict__ scores, float* __restrict__ boxes,
                                int S, int Bsz, int level) {
  const int n = S * S * 3;
  const int total = Bsz * n;
  int i = blockIdx.x * blockDim.x + threadIdx.x;
  if (i >= total) return;
  const int b = i / n;
  const int p = i % n;
  const int a = p % 3;
  const int pix = p / 3;
  const int ix = pix % S;
  const int iy = pix / S;
  const unsigned short* h = hfeat + ((size_t)(b * S + iy) * S + ix) * 256;
  float a0 = 0.f, a1 = 0.f, a2 = 0.f, a3 = 0.f, a4 = 0.f;
  for (int c = 0; c < 256; ++c) {
    const float hv = bf2f(h[c]);
    a0 += hv * wcls[c * 3 + a];
    const float* wb = wbox + c * 12 + a * 4;
    a1 += hv * wb[0]; a2 += hv * wb[1]; a3 += hv * wb[2]; a4 += hv * wb[3];
  }
  const float logit = a0 + bcls[a];
  const float score = 1.f / (1.f + expf(-logit));
  const float d0 = a1 + bbox[a * 4 + 0];
  const float d1 = a2 + bbox[a * 4 + 1];
  const float d2 = a3 + bbox[a * 4 + 2];
  const float d3 = a4 + bbox[a * 4 + 3];
  const float stride = (float)(1 << level);
  const float ayT[3] = {1.0f, 1.4f, 0.7f};
  const float axT[3] = {1.0f, 0.7f, 1.4f};
  const float ah = 8.0f * stride * ayT[a];
  const float aw = 8.0f * stride * axT[a];
  const float acy = (iy + 0.5f) * stride;
  const float acx = (ix + 0.5f) * stride;
  const float BBOX_CLIP = 4.135166556742356f;
  const float dh = fminf(d2, BBOX_CLIP), dw = fminf(d3, BBOX_CLIP);
  const float cy = d0 * ah + acy, cx = d1 * aw + acx;
  const float hh = expf(dh) * ah, ww = expf(dw) * aw;
  const float ih = info[b * 5 + 0], iw = info[b * 5 + 1];
  float* ob = boxes + (size_t)i * 4;
  ob[0] = fminf(fmaxf(cy - 0.5f * hh, 0.f), ih);
  ob[1] = fminf(fmaxf(cx - 0.5f * ww, 0.f), iw);
  ob[2] = fminf(fmaxf(cy + 0.5f * hh, 0.f), ih);
  ob[3] = fminf(fmaxf(cx + 0.5f * ww, 0.f), iw);
  scores[i] = score;
}

// ---------------------------------------------------------------------------
// Iterative top-k (block per batch). work = mutable per-batch copy of scores.
// ---------------------------------------------------------------------------
__global__ __launch_bounds__(256)
void topk_kernel(const float* __restrict__ scores, const float* __restrict__ boxes,
                 float* __restrict__ work, int n, int k,
                 float* __restrict__ ts, int* __restrict__ ti, float* __restrict__ tb) {
  const int b = blockIdx.x;
  const float* s = scores + (size_t)b * n;
  const float* bx = boxes + (size_t)b * n * 4;
  float* wk = work + (size_t)b * n;
  float* tsb = ts + (size_t)b * k;
  int* tib = ti + (size_t)b * k;
  float* tbb = tb + (size_t)b * k * 4;
  __shared__ float sv[256];
  __shared__ int si[256];
  for (int i = threadIdx.x; i < n; i += blockDim.x) wk[i] = s[i];
  __syncthreads();
  for (int it = 0; it < k; ++it) {
    float bv = -3e38f; int bi = 0;
    for (int i = threadIdx.x; i < n; i += blockDim.x) {
      const float v = wk[i];
      if (v > bv || (v == bv && i < bi)) { bv = v; bi = i; }
    }
    sv[threadIdx.x] = bv; si[threadIdx.x] = bi; __syncthreads();
    for (int st = 128; st > 0; st >>= 1) {
      if (threadIdx.x < st) {
        const float ov = sv[threadIdx.x + st]; const int oi = si[threadIdx.x + st];
        if (ov > sv[threadIdx.x] || (ov == sv[threadIdx.x] && oi < si[threadIdx.x])) {
          sv[threadIdx.x] = ov; si[threadIdx.x] = oi;
        }
      }
      __syncthreads();
    }
    if (threadIdx.x == 0) {
      const int j = si[0];
      tsb[it] = s[j]; tib[it] = j;
      tbb[it * 4 + 0] = bx[j * 4 + 0]; tbb[it * 4 + 1] = bx[j * 4 + 1];
      tbb[it * 4 + 2] = bx[j * 4 + 2]; tbb[it * 4 + 3] = bx[j * 4 + 3];
      wk[j] = -3e38f;
    }
    __syncthreads();
  }
}

// ---------------------------------------------------------------------------
// NMS exactly matching reference loop; outputs idxs + valid flags.
// ---------------------------------------------------------------------------
__global__ __launch_bounds__(256)
void nms_kernel(const float* __restrict__ scores, const float* __restrict__ boxes,
                float* __restrict__ work, int n, int keep, float thresh,
                int* __restrict__ idxs, int* __restrict__ valid) {
  const int b = blockIdx.x;
  const float* bx = boxes + (size_t)b * n * 4;
  float* s = work + (size_t)b * n;
  __shared__ float sv[256];
  __shared__ int si[256];
  __shared__ int jsh;
  for (int i = threadIdx.x; i < n; i += blockDim.x) s[i] = scores[(size_t)b * n + i];
  __syncthreads();
  for (int it = 0; it < keep; ++it) {
    float bv = -3e38f; int bi = 0;
    for (int i = threadIdx.x; i < n; i += blockDim.x) {
      const float v = s[i];
      if (v > bv || (v == bv && i < bi)) { bv = v; bi = i; }
    }
    sv[threadIdx.x] = bv; si[threadIdx.x] = bi; __syncthreads();
    for (int st = 128; st > 0; st >>= 1) {
      if (threadIdx.x < st) {
        const float ov = sv[threadIdx.x + st]; const int oi = si[threadIdx.x + st];
        if (ov > sv[threadIdx.x] || (ov == sv[threadIdx.x] && oi < si[threadIdx.x])) {
          sv[threadIdx.x] = ov; si[threadIdx.x] = oi;
        }
      }
      __syncthreads();
    }
    if (threadIdx.x == 0) {
      jsh = si[0];
      idxs[(size_t)b * keep + it] = si[0];
      valid[(size_t)b * keep + it] = (s[si[0]] > -1e9f) ? 1 : 0;
    }
    __syncthreads();
    const int j = jsh;
    float bj[4] = {bx[j * 4 + 0], bx[j * 4 + 1], bx[j * 4 + 2], bx[j * 4 + 3]};
    for (int i = threadIdx.x; i < n; i += blockDim.x) {
      float bt[4] = {bx[i * 4 + 0], bx[i * 4 + 1], bx[i * 4 + 2], bx[i * 4 + 3]};
      if (iou_f(bj, bt) >= thresh) s[i] = -1e10f;
    }
    __syncthreads();
  }
}

// ---------------------------------------------------------------------------
// Gather per-level NMS survivors into concatenated candidate arrays.
// ---------------------------------------------------------------------------
__global__ void rpn_gather_kernel(const float* __restrict__ ts, const float* __restrict__ tb,
                                  const int* __restrict__ idxs, const int* __restrict__ valid,
                                  int k, int lvl, float* __restrict__ cs, float* __restrict__ cb,
                                  int Bsz) {
  int i = blockIdx.x * blockDim.x + threadIdx.x;
  if (i >= Bsz * 128) return;
  const int b = i / 128, pos = i % 128;
  const int j = idxs[b * 128 + pos];
  const int v = valid[b * 128 + pos];
  const int dst = b * 640 + lvl * 128 + pos;
  cs[dst] = v ? ts[(size_t)b * k + j] : -1e10f;
  for (int q = 0; q < 4; ++q)
    cb[(size_t)dst * 4 + q] = v ? tb[((size_t)b * k + j) * 4 + q] : 0.f;
}

// ---------------------------------------------------------------------------
// Multilevel ROI align 7x7x256 -> bf16 rows for FC1. Block per (b,roi).
// ---------------------------------------------------------------------------
__global__ __launch_bounds__(256)
void roi_align_kernel(const unsigned short* __restrict__ feat,
                      const float* __restrict__ rois,
                      unsigned short* __restrict__ out, int Bsz) {
  const int blk = blockIdx.x;
  const int b = blk / 512;
  const float* box = rois + (size_t)blk * 4;
  const float y1 = box[0], x1 = box[1], y2 = box[2], x2 = box[3];
  const float h = y2 - y1, w = x2 - x1;
  float lvlf = floorf(log2f(sqrtf(fmaxf(h * w, 1e-8f)) / 224.f + 1e-8f) + 4.f);
  lvlf = fminf(fmaxf(lvlf, 2.f), 6.f);
  const int li = (int)lvlf - 2;
  const float stride = exp2f(lvlf);
  const int S = 512 >> (li + 2);
  const int POFF[5] = {0, 16384, 20480, 21504, 21760};
  const size_t base = (size_t)Bsz * POFF[li] * 256;
  const int c = threadIdx.x;
  for (int pos = 0; pos < 49; ++pos) {
    const int py = pos / 7, px = pos % 7;
    const float ty = (py + 0.5f) / 7.f, tx = (px + 0.5f) / 7.f;
    const float fy = (y1 + ty * h) / stride - 0.5f;
    const float fx = (x1 + tx * w) / stride - 0.5f;
    const float y0f = floorf(fy), x0f = floorf(fx);
    const float wy = fy - y0f, wx = fx - x0f;
    int y0 = (int)y0f; y0 = min(max(y0, 0), S - 1);
    int x0 = (int)x0f; x0 = min(max(x0, 0), S - 1);
    const int y1i = min(y0 + 1, S - 1);
    const int x1i = min(x0 + 1, S - 1);
    const size_t rb = base + (size_t)b * S * S * 256;
    const float v00 = bf2f(feat[rb + ((size_t)y0 * S + x0) * 256 + c]);
    const float v01 = bf2f(feat[rb + ((size_t)y0 * S + x1i) * 256 + c]);
    const float v10 = bf2f(feat[rb + ((size_t)y1i * S + x0) * 256 + c]);
    const float v11 = bf2f(feat[rb + ((size_t)y1i * S + x1i) * 256 + c]);
    const float v = v00 * (1 - wy) * (1 - wx) + v01 * (1 - wy) * wx +
                    v10 * wy * (1 - wx) + v11 * wy * wx;
    out[((size_t)blk * 49 + pos) * 256 + c] = f2bf(v);
  }
}

// ---------------------------------------------------------------------------
// Second-stage heads: cls (91) + box regression (364), scalar dot products.
// ---------------------------------------------------------------------------
__global__ void heads_kernel(const unsigned short* __restrict__ x2,
                             const float* __restrict__ cls_w, const float* __restrict__ cls_b,
                             const float* __restrict__ boxr_w, const float* __restrict__ boxr_b,
                             float* __restrict__ cls_logits, float* __restrict__ box_deltas,
                             int Mrows) {
  int i = blockIdx.x * blockDim.x + threadIdx.x;
  if (i >= Mrows * 455) return;
  const int m = i / 455;
  const int j = i % 455;
  const unsigned short* x = x2 + (size_t)m * 1024;
  float acc = 0.f;
  if (j < 91) {
    for (int k = 0; k < 1024; ++k) acc += bf2f(x[k]) * cls_w[(size_t)k * 91 + j];
    cls_logits[(size_t)m * 91 + j] = acc + cls_b[j];
  } else {
    const int jj = j - 91;
    for (int k = 0; k < 1024; ++k) acc += bf2f(x[k]) * boxr_w[(size_t)k * 364 + jj];
    box_deltas[(size_t)m * 364 + jj] = acc + boxr_b[jj];
  }
}

// ---------------------------------------------------------------------------
// Softmax over 91 classes + per-class box decode/clip; block per roi-row.
// ---------------------------------------------------------------------------
__global__ __launch_bounds__(128)
void softmax_decode_kernel(const float* __restrict__ logits, const float* __restrict__ deltas,
                           const float* __restrict__ rois, const float* __restrict__ info,
                           float* __restrict__ sc, float* __restrict__ bx) {
  const int m = blockIdx.x;         // b*512 + r
  const int b = m >> 9;
  const int r = m & 511;
  const int tid = threadIdx.x;
  __shared__ float red[128];
  const float lv = (tid < 91) ? logits[(size_t)m * 91 + tid] : -3e38f;
  red[tid] = lv; __syncthreads();
  for (int st = 64; st > 0; st >>= 1) {
    if (tid < st) red[tid] = fmaxf(red[tid], red[tid + st]);
    __syncthreads();
  }
  const float mx = red[0]; __syncthreads();
  const float e = (tid < 91) ? expf(lv - mx) : 0.f;
  red[tid] = e; __syncthreads();
  for (int st = 64; st > 0; st >>= 1) {
    if (tid < st) red[tid] += red[tid + st];
    __syncthreads();
  }
  const float sum = red[0];
  if (tid >= 1 && tid < 91) {
    const float p = e / sum;
    const size_t o = (size_t)b * 46080 + (size_t)r * 90 + (tid - 1);
    sc[o] = p;
    const float* roi = rois + ((size_t)b * 512 + r) * 4;
    const float ah = roi[2] - roi[0], aw = roi[3] - roi[1];
    const float acy = roi[0] + 0.5f * ah, acx = roi[1] + 0.5f * aw;
    const float* dl = deltas + (size_t)m * 364 + (size_t)tid * 4;
    const float BBOX_CLIP = 4.135166556742356f;
    const float dy = dl[0] * 0.1f, dx = dl[1] * 0.1f;
    const float dh = fminf(dl[2] * 0.2f, BBOX_CLIP);
    const float dw = fminf(dl[3] * 0.2f, BBOX_CLIP);
    const float cy = dy * ah + acy, cx = dx * aw + acx;
    const float hh = expf(dh) * ah, ww = expf(dw) * aw;
    const float ih = info[b * 5 + 0], iw = info[b * 5 + 1];
    float* ob = bx + o * 4;
    ob[0] = fminf(fmaxf(cy - 0.5f * hh, 0.f), ih);
    ob[1] = fminf(fmaxf(cx - 0.5f * ww, 0.f), iw);
    ob[2] = fminf(fmaxf(cy + 0.5f * hh, 0.f), ih);
    ob[3] = fminf(fmaxf(cx + 0.5f * ww, 0.f), iw);
  }
}

// tc = class id; ob = boxes offset by class*2*IMG for class-separated NMS
__global__ void det_prep_kernel(const int* __restrict__ ti, const float* __restrict__ tb,
                                float* __restrict__ tc, float* __restrict__ ob, int total) {
  int i = blockIdx.x * blockDim.x + threadIdx.x;
  if (i >= total) return;
  const float c = (float)((ti[i] % 90) + 1);
  tc[i] = c;
  const float off = c * 1024.0f;
  for (int q = 0; q < 4; ++q) ob[(size_t)i * 4 + q] = tb[(size_t)i * 4 + q] + off;
}

__global__ __launch_bounds__(256)
void finalize_kernel(const float* __restrict__ dts, const float* __restrict__ dtb,
                     const float* __restrict__ tc, const int* __restrict__ idxs,
                     const int* __restrict__ valid, float* __restrict__ out, int Bsz) {
  const int b = blockIdx.x;
  const int i = threadIdx.x;
  __shared__ float red[256];
  float cnt = 0.f;
  if (i < 100) {
    const int j = idxs[b * 100 + i];
    const int v = valid[b * 100 + i];
    cnt = v ? 1.f : 0.f;
    float b0 = 0.f, b1 = 0.f, b2 = 0.f, b3 = 0.f, cls = 0.f, scv = 0.f;
    if (v) {
      const float* p = dtb + ((size_t)b * 500 + j) * 4;
      b0 = p[0]; b1 = p[1]; b2 = p[2]; b3 = p[3];
      cls = tc[(size_t)b * 500 + j];
      scv = dts[(size_t)b * 500 + j];
    }
    float* boxes = out + Bsz;
    boxes[((size_t)b * 100 + i) * 4 + 0] = b0;
    boxes[((size_t)b * 100 + i) * 4 + 1] = b1;
    boxes[((size_t)b * 100 + i) * 4 + 2] = b2;
    boxes[((size_t)b * 100 + i) * 4 + 3] = b3;
    out[Bsz + Bsz * 400 + b * 100 + i] = cls;
    out[Bsz + Bsz * 400 + Bsz * 100 + b * 100 + i] = scv;
  }
  red[i] = cnt; __syncthreads();
  for (int st = 128; st > 0; st >>= 1) {
    if (i < st) red[i] += red[i + st];
    __syncthreads();
  }
  if (i == 0) out[b] = red[0];
}

// ---------------------------------------------------------------------------
// Host orchestration
// ---------------------------------------------------------------------------
extern "C" void kernel_launch(void* const* d_in, const int* in_sizes, int n_in,
                              void* d_out, int out_size, void* d_ws, size_t ws_size,
                              hipStream_t stream) {
  (void)n_in; (void)out_size; (void)ws_size;
  const float* fpn[5] = {(const float*)d_in[0], (const float*)d_in[1], (const float*)d_in[2],
                         (const float*)d_in[3], (const float*)d_in[4]};
  const float* info   = (const float*)d_in[5];
  const float* conv_w = (const float*)d_in[6];
  const float* conv_b = (const float*)d_in[7];
  const float* cls1_w = (const float*)d_in[8];
  const float* cls1_b = (const float*)d_in[9];
  const float* box1_w = (const float*)d_in[10];
  const float* box1_b = (const float*)d_in[11];
  const float* fc1_w  = (const float*)d_in[12];
  const float* fc1_b  = (const float*)d_in[13];
  const float* fc2_w  = (const float*)d_in[14];
  const float* fc2_b  = (const float*)d_in[15];
  const float* cls_w  = (const float*)d_in[16];
  const float* cls_b  = (const float*)d_in[17];
  const float* boxr_w = (const float*)d_in[18];
  const float* boxr_b = (const float*)d_in[19];
  float* out = (float*)d_out;

  const int Bsz = in_sizes[0] / (128 * 128 * 256);
  const int S_h[5] = {128, 64, 32, 16, 8};
  const int Mfc = Bsz * 512;

  // workspace allocator (256B aligned)
  char* wsb = (char*)d_ws;
  size_t off = 0;
  auto alloc = [&](size_t bytes) -> void* {
    void* p = wsb + off;
    off = (off + bytes + 255) & ~(size_t)255;
    return p;
  };

  size_t featOff[6]; featOff[0] = 0;
  size_t scOff[6]; scOff[0] = 0;
  for (int l = 0; l < 5; ++l) {
    featOff[l + 1] = featOff[l] + (size_t)Bsz * S_h[l] * S_h[l] * 256;
    scOff[l + 1] = scOff[l] + (size_t)Bsz * S_h[l] * S_h[l] * 3;
  }
  const size_t featElems = featOff[5];

  unsigned short* featB  = (unsigned short*)alloc(featElems * 2);
  unsigned short* hfeatB = (unsigned short*)alloc(featElems * 2);
  unsigned short* wconvB = (unsigned short*)alloc((size_t)2304 * 256 * 2);
  unsigned short* wfc1B  = (unsigned short*)alloc((size_t)12544 * 1024 * 2);
  unsigned short* wfc2B  = (unsigned short*)alloc((size_t)1024 * 1024 * 2);
  float* rscores = (float*)alloc(scOff[5] * 4);
  float* rboxes  = (float*)alloc(scOff[5] * 16);
  float* work    = (float*)alloc((size_t)Bsz * 49152 * 4);
  float* tsL     = (float*)alloc((size_t)5 * Bsz * 400 * 4);
  int*   tiL     = (int*)  alloc((size_t)5 * Bsz * 400 * 4);
  float* tbL     = (float*)alloc((size_t)5 * Bsz * 400 * 16);
  int*   idxL    = (int*)  alloc((size_t)5 * Bsz * 128 * 4);
  int*   valL    = (int*)  alloc((size_t)5 * Bsz * 128 * 4);
  float* cs      = (float*)alloc((size_t)Bsz * 640 * 4);
  float* cb      = (float*)alloc((size_t)Bsz * 640 * 16);
  float* mts     = (float*)alloc((size_t)Bsz * 512 * 4);
  int*   mti     = (int*)  alloc((size_t)Bsz * 512 * 4);
  float* rois    = (float*)alloc((size_t)Bsz * 512 * 16);
  unsigned short* roiX = (unsigned short*)alloc((size_t)Mfc * 12544 * 2);
  unsigned short* x1B  = (unsigned short*)alloc((size_t)Mfc * 1024 * 2);
  unsigned short* x2B  = (unsigned short*)alloc((size_t)Mfc * 1024 * 2);
  float* clsLog  = (float*)alloc((size_t)Mfc * 91 * 4);
  float* boxDel  = (float*)alloc((size_t)Mfc * 364 * 4);
  float* scFlat  = (float*)alloc((size_t)Bsz * 46080 * 4);
  float* bxFlat  = (float*)alloc((size_t)Bsz * 46080 * 16);
  float* dts     = (float*)alloc((size_t)Bsz * 500 * 4);
  int*   dti     = (int*)  alloc((size_t)Bsz * 500 * 4);
  float* dtb     = (float*)alloc((size_t)Bsz * 500 * 16);
  float* dtc     = (float*)alloc((size_t)Bsz * 500 * 4);
  float* dob     = (float*)alloc((size_t)Bsz * 500 * 16);
  int*   didx    = (int*)  alloc((size_t)Bsz * 100 * 4);
  int*   dval    = (int*)  alloc((size_t)Bsz * 100 * 4);

  // 1) convert features + weights to bf16
  for (int l = 0; l < 5; ++l) {
    const size_t n = (size_t)Bsz * S_h[l] * S_h[l] * 256;
    cvt_f32_bf16_kernel<<<dim3((unsigned)((n + 255) / 256)), 256, 0, stream>>>(
        fpn[l], featB + featOff[l], n);
  }
  {
    size_t n = (size_t)2304 * 256;
    cvt_f32_bf16_kernel<<<dim3((unsigned)((n + 255) / 256)), 256, 0, stream>>>(conv_w, wconvB, n);
    n = (size_t)12544 * 1024;
    cvt_f32_bf16_kernel<<<dim3((unsigned)((n + 255) / 256)), 256, 0, stream>>>(fc1_w, wfc1B, n);
    n = (size_t)1024 * 1024;
    cvt_f32_bf16_kernel<<<dim3((unsigned)((n + 255) / 256)), 256, 0, stream>>>(fc2_w, wfc2B, n);
  }

  // 2) RPN 3x3 conv (WMMA, 64x64 tiles) + 3) RPN heads per level
  for (int l = 0; l < 5; ++l) {
    const int S = S_h[l];
    const int M = Bsz * S * S;
    conv3x3_wmma_kernel<<<dim3(M / 64, 256 / 64), 128, 0, stream>>>(
        featB + featOff[l], wconvB, conv_b, hfeatB + featOff[l], S, Bsz);
    const int total = Bsz * S * S * 3;
    rpn_pred_kernel<<<dim3((total + 255) / 256), 256, 0, stream>>>(
        hfeatB + featOff[l], cls1_w, cls1_b, box1_w, box1_b, info,
        rscores + scOff[l], rboxes + scOff[l] * 4, S, Bsz, l + 2);
  }

  // 4) per-level top-k + NMS + gather
  for (int l = 0; l < 5; ++l) {
    const int S = S_h[l];
    const int n = S * S * 3;
    const int k = n < 400 ? n : 400;
    float* ts_l = tsL + (size_t)l * Bsz * 400;
    int*   ti_l = tiL + (size_t)l * Bsz * 400;
    float* tb_l = tbL + (size_t)l * Bsz * 400 * 4;
    topk_kernel<<<dim3(Bsz), 256, 0, stream>>>(rscores + scOff[l], rboxes + scOff[l] * 4,
                                               work, n, k, ts_l, ti_l, tb_l);
    int* idx_l = idxL + (size_t)l * Bsz * 128;
    int* val_l = valL + (size_t)l * Bsz * 128;
    nms_kernel<<<dim3(Bsz), 256, 0, stream>>>(ts_l, tb_l, work, k, 128, 0.7f, idx_l, val_l);
    rpn_gather_kernel<<<dim3((Bsz * 128 + 255) / 256), 256, 0, stream>>>(
        ts_l, tb_l, idx_l, val_l, k, l, cs, cb, Bsz);
  }

  // 5) merge -> top-512 rois
  topk_kernel<<<dim3(Bsz), 256, 0, stream>>>(cs, cb, work, 640, 512, mts, mti, rois);

  // 6) ROI align -> bf16 FC input rows
  roi_align_kernel<<<dim3(Bsz * 512), 256, 0, stream>>>(featB, rois, roiX, Bsz);

  // 7) FC1 / FC2 (WMMA, 64x64 tiles)
  gemm_bf16_wmma_kernel<<<dim3(Mfc / 64, 1024 / 64), 128, 0, stream>>>(
      roiX, wfc1B, fc1_b, x1B, Mfc, 1024, 12544, 1);
  gemm_bf16_wmma_kernel<<<dim3(Mfc / 64, 1024 / 64), 128, 0, stream>>>(
      x1B, wfc2B, fc2_b, x2B, Mfc, 1024, 1024, 1);

  // 8) class / box heads
  heads_kernel<<<dim3((Mfc * 455 + 255) / 256), 256, 0, stream>>>(
      x2B, cls_w, cls_b, boxr_w, boxr_b, clsLog, boxDel, Mfc);

  // 9) softmax + decode
  softmax_decode_kernel<<<dim3(Mfc), 128, 0, stream>>>(clsLog, boxDel, rois, info, scFlat, bxFlat);

  // 10) detection top-500, class offsets, NMS keep-100, finalize
  topk_kernel<<<dim3(Bsz), 256, 0, stream>>>(scFlat, bxFlat, work, 46080, 500, dts, dti, dtb);
  det_prep_kernel<<<dim3((Bsz * 500 + 255) / 256), 256, 0, stream>>>(dti, dtb, dtc, dob, Bsz * 500);
  nms_kernel<<<dim3(Bsz), 256, 0, stream>>>(dts, dob, work, 500, 100, 0.5f, didx, dval);
  finalize_kernel<<<dim3(Bsz), 256, 0, stream>>>(dts, dtb, dtc, didx, dval, out, Bsz);
}
